// stau_24661702213808
// MI455X (gfx1250) — matrix operand order repaired and checked by reading, not compiled, not run on hardware
//
#include <hip/hip_runtime.h>
#include <cstddef>
#include <cstdint>

// ---------------------------------------------------------------------------
// Problem constants (from the reference)
// ---------------------------------------------------------------------------
constexpr int    Hdim  = 256;
constexpr int    BATCH = 8192;
constexpr int    NLAY  = 4;
constexpr int    TAUc  = 8;
constexpr size_t BH    = (size_t)BATCH * Hdim;   // 2,097,152 floats (8 MB)

typedef __attribute__((ext_vector_type(16))) __bf16 v16bf;
typedef __attribute__((ext_vector_type(8)))  float  v8f;

// ---------------------------------------------------------------------------
// Convert one 256x256 f32 weight matrix to bf16 (row-major, same layout).
// ---------------------------------------------------------------------------
__global__ __launch_bounds__(256)
void convert_w_kernel(const float* __restrict__ W, __bf16* __restrict__ Wb)
{
    const int i = (blockIdx.x * 256 + threadIdx.x) * 4;
    const float4 w = *reinterpret_cast<const float4*>(W + i);
    union { __bf16 h[4]; unsigned long long u; } cv;
    cv.h[0] = (__bf16)w.x; cv.h[1] = (__bf16)w.y;
    cv.h[2] = (__bf16)w.z; cv.h[3] = (__bf16)w.w;
    *reinterpret_cast<unsigned long long*>(Wb + i) = cv.u;
}

// ---------------------------------------------------------------------------
// GEMM: C[M,N] = act(A[M,K=256] @ W[N,K=256]^T + bias[N]),  M=8192, N=256
// bf16 WMMA with f32 accumulate. Block = 256 threads = 8 waves,
// block tile = 128 rows x 128 cols; wave tile = 16 rows x 128 cols.
// Weight tile (bf16, pre-converted) staged into LDS with async copies.
// __launch_bounds__(256, 1): one block per WGP (it owns 64KB LDS anyway),
// so the compiler may keep acc(64) + A-strip(64) + B-frags(64) VGPRs live.
// ---------------------------------------------------------------------------
__global__ __launch_bounds__(256, 1)
void gemm_bf16_kernel(const float* __restrict__ A,
                      const __bf16* __restrict__ Wb,
                      const float* __restrict__ bias,
                      float* __restrict__ C,
                      int act)
{
    __shared__ __bf16 Wl[128 * 256];   // [n_local][k] bf16, 64 KB

    const int tid = threadIdx.x;
    const int n0  = blockIdx.y * 128;

    // ---- async stage 128x256 bf16 weight tile: global -> LDS --------------
    {
        const unsigned lds_base = (unsigned)(uintptr_t)(&Wl[0]);
        const char*    gsrc     = (const char*)(Wb + (size_t)n0 * 256);
#pragma unroll
        for (int j = 0; j < 16; ++j) {
            const unsigned chunk = (unsigned)(j * 256 + tid) * 16u; // byte off
            const unsigned           ldsa = lds_base + chunk;
            const unsigned long long ga   = (unsigned long long)(uintptr_t)(gsrc + chunk);
            asm volatile("global_load_async_to_lds_b128 %0, %1, off"
                         :: "v"(ldsa), "v"(ga)
                         : "memory");
        }
        asm volatile("s_wait_asynccnt 0" ::: "memory");
    }
    __syncthreads();

    const int wave = tid >> 5;
    const int lane = tid & 31;
    const int m0   = blockIdx.x * 128 + wave * 16;
    const int mrow = lane & 15;            // row (A frag) / col (B,D frags)
    const int koff = (lane >> 4) * 16;     // A-frag K sub-offset (contiguous K, k = 16h + s)
    const int kk   = (lane >> 4) * 16;     // B-frag K sub-offset (contiguous K, k = 16h + s)

    const v8f vzero = {0.f, 0.f, 0.f, 0.f, 0.f, 0.f, 0.f, 0.f};
    v8f acc[8];
#pragma unroll
    for (int t = 0; t < 8; ++t) acc[t] = vzero;

    const float* Arow = A + (size_t)(m0 + mrow) * Hdim;

    // ---- preload the whole 16x256 A strip as bf16 WMMA fragments ----------
    // All 64 float2 global loads issue as one clause, then one conversion burst.
    // A fragment 16x32 bf16 (contiguous K, matching B):
    //  lane<16:  M=lane,    K = {0..7}  in v0..3, {8..15} in v4..7
    //  lane>=16: M=lane-16, K = {16..23} in v0..3, {24..31} in v4..7
    v16bf afrag[8];
#pragma unroll
    for (int kt = 0; kt < 8; ++kt) {
        const int k0 = kt * 32;
        v16bf a;
#pragma unroll
        for (int v = 0; v < 4; ++v) {
            const float2 p =
                *reinterpret_cast<const float2*>(Arow + k0 + koff + 2 * v);
            a[2 * v]     = (__bf16)p.x;
            a[2 * v + 1] = (__bf16)p.y;
            const float2 q =
                *reinterpret_cast<const float2*>(Arow + k0 + 8 + koff + 2 * v);
            a[8 + 2 * v]     = (__bf16)q.x;
            a[8 + 2 * v + 1] = (__bf16)q.y;
        }
        afrag[kt] = a;
    }

    // ---- K loop: per step one ds_load clause, then 8 back-to-back WMMAs ---
#pragma unroll
    for (int kt = 0; kt < 8; ++kt) {
        const int k0 = kt * 32;
        v16bf bf[8];
#pragma unroll
        for (int nt = 0; nt < 8; ++nt) {
            // B fragment 32x16 bf16: lane<16 -> N=lane, K=kk..kk+15 contiguous
            bf[nt] = *reinterpret_cast<const v16bf*>(
                &Wl[(nt * 16 + mrow) * 256 + k0 + kk]);
        }
#pragma unroll
        for (int nt = 0; nt < 8; ++nt) {
            acc[nt] = __builtin_amdgcn_wmma_f32_16x16x32_bf16(
                false, afrag[kt], false, bf[nt], (short)0, acc[nt],
                false, false);
        }
    }

    // ---- epilogue: D layout (f32 16x16): VGPR r -> M = r + (lane>=16 ? 8:0)
    const int rbase = (lane >> 4) * 8;
#pragma unroll
    for (int nt = 0; nt < 8; ++nt) {
        const int   col = n0 + nt * 16 + mrow;
        const float bv  = bias[col];
#pragma unroll
        for (int r = 0; r < 8; ++r) {
            float v = acc[nt][r] + bv;
            if (act == 1) v = tanhf(v);
            C[(size_t)(m0 + rbase + r) * Hdim + col] = v;
        }
    }
}

// ---------------------------------------------------------------------------
// Attention + gating fusion. One wave32 per batch row b; lane owns 8 h-elems.
// ---------------------------------------------------------------------------
__device__ __forceinline__ void ld8(const float* __restrict__ p, float* r) {
    const float4 a = reinterpret_cast<const float4*>(p)[0];
    const float4 b = reinterpret_cast<const float4*>(p)[1];
    r[0] = a.x; r[1] = a.y; r[2] = a.z; r[3] = a.w;
    r[4] = b.x; r[5] = b.y; r[6] = b.z; r[7] = b.w;
}
__device__ __forceinline__ void st8(float* __restrict__ p, const float* r) {
    reinterpret_cast<float4*>(p)[0] = make_float4(r[0], r[1], r[2], r[3]);
    reinterpret_cast<float4*>(p)[1] = make_float4(r[4], r[5], r[6], r[7]);
}
__device__ __forceinline__ float wave_sum(float p) {
#pragma unroll
    for (int off = 16; off > 0; off >>= 1) p += __shfl_xor(p, off, 32);
    return p;
}
__device__ __forceinline__ float sigmoidf(float x) {
    return 1.f / (1.f + expf(-x));
}

__global__ __launch_bounds__(256)
void attn_fuse_kernel(int layer,
                      const float* __restrict__ t_att,
                      const float* __restrict__ s_att,
                      const float* __restrict__ S_hist,
                      const float* __restrict__ s_next_g,
                      const float* __restrict__ t_next_g,
                      float* __restrict__ T_fus,
                      float* __restrict__ S_fus)
{
    const int    lane   = threadIdx.x & 31;
    const int    b      = blockIdx.x * 8 + (threadIdx.x >> 5);
    const size_t rowoff = (size_t)b * Hdim + lane * 8;
    const size_t lt     = (size_t)layer * TAUc * BH;
    const float  scale  = 0.0625f;   // 1/sqrt(256)

    float sn[8], tn[8], Sc[8];
    ld8(s_next_g + rowoff, sn);
    ld8(t_next_g + rowoff, tn);
    ld8(S_hist + (size_t)layer * BH + rowoff, Sc);   // current S = S_hist[layer]

    // ---- temporal attention: keys = [s_att[i,1:], S], values = t_att[i] ----
    float wk[8];
    float m = -1e30f;
#pragma unroll
    for (int k = 0; k < 8; ++k) {
        // prefetch the t_att value row we will stream in the trend loop
        __builtin_prefetch(t_att + lt + (size_t)k * BH + rowoff, 0, 1);
        float p = 0.f;
        if (k < 7) {
            float kv[8];
            ld8(s_att + lt + (size_t)(k + 1) * BH + rowoff, kv);
#pragma unroll
            for (int j = 0; j < 8; ++j) p += kv[j] * sn[j];
        } else {
#pragma unroll
            for (int j = 0; j < 8; ++j) p += Sc[j] * sn[j];
        }
        p     = wave_sum(p) * scale;
        wk[k] = p;
        m     = fmaxf(m, p);
    }
    float Z = 0.f;
#pragma unroll
    for (int k = 0; k < 8; ++k) { wk[k] = expf(wk[k] - m); Z += wk[k]; }
    const float invZ = 1.f / Z;

    float tt[8] = {0.f, 0.f, 0.f, 0.f, 0.f, 0.f, 0.f, 0.f};
    float Tl[8];
#pragma unroll
    for (int k = 0; k < 8; ++k) {
        float tv[8];
        ld8(t_att + lt + (size_t)k * BH + rowoff, tv);
#pragma unroll
        for (int j = 0; j < 8; ++j) tt[j] += wk[k] * tv[j];
        if (k == 7) {
#pragma unroll
            for (int j = 0; j < 8; ++j) Tl[j] = tv[j];
        }
    }
    float tf[8];
#pragma unroll
    for (int j = 0; j < 8; ++j) {
        const float g = sigmoidf(tn[j]);
        tf[j] = Tl[j] * g + (1.f - g) * (tt[j] * invZ);
    }
    st8(T_fus + rowoff, tf);

    // ---- spatial attention over rolling buffers (zero-padded to theta=8) ---
    // keys: t_sp = [0 x (7-layer), t_att[0,-1] .. t_att[layer,-1]]
    // vals: s_sp = [0 x (7-layer), S_0 .. S_layer]
    const int nreal = layer + 1;
    float sc2[4], e2[4];
#pragma unroll 4
    for (int j2 = 0; j2 < nreal; ++j2) {
        float kv[8];
        ld8(t_att + ((size_t)j2 * TAUc + 7) * BH + rowoff, kv);
        float p = 0.f;
#pragma unroll
        for (int j = 0; j < 8; ++j) p += kv[j] * tn[j];
        sc2[j2] = wave_sum(p) * scale;
    }
    float m2 = 0.f;   // zero keys score exactly 0
#pragma unroll 4
    for (int j2 = 0; j2 < nreal; ++j2) m2 = fmaxf(m2, sc2[j2]);
    float Z2 = (float)(7 - layer) * expf(-m2);   // zero-key softmax mass
#pragma unroll 4
    for (int j2 = 0; j2 < nreal; ++j2) { e2[j2] = expf(sc2[j2] - m2); Z2 += e2[j2]; }
    const float invZ2 = 1.f / Z2;

    float st[8] = {0.f, 0.f, 0.f, 0.f, 0.f, 0.f, 0.f, 0.f};
#pragma unroll 4
    for (int j2 = 0; j2 < nreal; ++j2) {
        float sv[8];
        ld8(S_hist + (size_t)j2 * BH + rowoff, sv);
        const float wj = e2[j2] * invZ2;
#pragma unroll
        for (int j = 0; j < 8; ++j) st[j] += wj * sv[j];
    }
    float sf[8];
#pragma unroll
    for (int j = 0; j < 8; ++j) {
        const float g = sigmoidf(sn[j]);
        sf[j] = Sc[j] * g + (1.f - g) * st[j];
    }
    st8(S_fus + rowoff, sf);
}

// ---------------------------------------------------------------------------
// S_{i+1} = sigmoid(s_g) * s_s + (1 - sigmoid(s_g)) * t_s
// ---------------------------------------------------------------------------
__global__ __launch_bounds__(256)
void combine_kernel(const float* __restrict__ s_g,
                    const float* __restrict__ s_s,
                    const float* __restrict__ t_s,
                    float* __restrict__ S_out)
{
    const size_t i = (size_t)blockIdx.x * 256 + threadIdx.x;
    const float  g = 1.f / (1.f + expf(-s_g[i]));
    S_out[i] = g * s_s[i] + (1.f - g) * t_s[i];
}

// ---------------------------------------------------------------------------
// Host-side orchestration
// ---------------------------------------------------------------------------
extern "C" void kernel_launch(void* const* d_in, const int* in_sizes, int n_in,
                              void* d_out, int out_size, void* d_ws, size_t ws_size,
                              hipStream_t stream)
{
    (void)in_sizes; (void)n_in; (void)out_size; (void)ws_size;

    const float* x       = (const float*)d_in[0];
    const float* t_att   = (const float*)d_in[1];
    const float* s_att   = (const float*)d_in[2];
    const float* embed_w = (const float*)d_in[3];
    const float* embed_b = (const float*)d_in[4];
    const float* wtn     = (const float*)d_in[5];
    const float* btn     = (const float*)d_in[6];
    const float* wsn     = (const float*)d_in[7];
    const float* bsn     = (const float*)d_in[8];
    const float* wt      = (const float*)d_in[9];
    const float* bt      = (const float*)d_in[10];
    const float* wsw     = (const float*)d_in[11];
    const float* bs      = (const float*)d_in[12];
    const float* out_w   = (const float*)d_in[13];
    const float* out_b   = (const float*)d_in[14];
    float*       out     = (float*)d_out;

    float* wsf    = (float*)d_ws;
    float* S_hist = wsf;                 // 4 * BH : S_0 .. S_3
    float* s_next = wsf + 4 * BH;
    float* t_next = wsf + 5 * BH;
    float* T_fus  = wsf + 6 * BH;
    float* S_fus  = wsf + 7 * BH;
    float* t_s    = wsf + 8 * BH;
    float* s_g    = wsf + 9 * BH;
    float* s_s    = wsf + 10 * BH;
    float* S_fin  = wsf + 11 * BH;
    __bf16* Wbf   = (__bf16*)(wsf + 12 * BH);   // 256*256 bf16 scratch

    const dim3 ggrid(BATCH / 128, Hdim / 128);   // (64, 2)
    const dim3 gblk(256);
    const dim3 cgrid(Hdim * Hdim / (256 * 4));   // 64 blocks per weight convert

#define RUN_GEMM(Aptr, Wptr, Bptr, Cptr, ACT)                                   \
    do {                                                                        \
        convert_w_kernel<<<cgrid, gblk, 0, stream>>>((Wptr), Wbf);              \
        gemm_bf16_kernel<<<ggrid, gblk, 0, stream>>>((Aptr), Wbf, (Bptr),       \
                                                     (Cptr), (ACT));            \
    } while (0)

    // embed: S_0 = x @ embed_w^T + embed_b
    RUN_GEMM(x, embed_w, embed_b, S_hist, 0);

    for (int i = 0; i < NLAY; ++i) {
        const float* Si     = S_hist + (size_t)i * BH;
        const float* T_last = t_att + ((size_t)i * TAUc + (TAUc - 1)) * BH;

        RUN_GEMM(Si, wsn + (size_t)i * Hdim * Hdim, bsn + (size_t)i * Hdim,
                 s_next, 0);
        RUN_GEMM(T_last, wtn + (size_t)i * Hdim * Hdim, btn + (size_t)i * Hdim,
                 t_next, 0);

        attn_fuse_kernel<<<dim3(BATCH / 8), gblk, 0, stream>>>(
            i, t_att, s_att, S_hist, s_next, t_next, T_fus, S_fus);

        // only the last third of Tc (t_s_) and the first/last thirds of Sc
        // (s_g, s_s_) are live in the reference -> 3 HxH GEMMs on slices.
        const size_t w3 = (size_t)i * 3 * Hdim * Hdim;
        const size_t b3 = (size_t)i * 3 * Hdim;
        RUN_GEMM(T_fus, wt + w3 + 2 * (size_t)Hdim * Hdim, bt + b3 + 2 * Hdim,
                 t_s, 0);
        RUN_GEMM(S_fus, wsw + w3, bs + b3, s_g, 0);
        RUN_GEMM(S_fus, wsw + w3 + 2 * (size_t)Hdim * Hdim, bs + b3 + 2 * Hdim,
                 s_s, 0);

        float* dst = (i < NLAY - 1) ? (S_hist + (size_t)(i + 1) * BH) : S_fin;
        combine_kernel<<<dim3((unsigned)(BH / 256)), gblk, 0, stream>>>(
            s_g, s_s, t_s, dst);
    }

    // out = tanh(S @ out_w^T + out_b)
    RUN_GEMM(S_fin, out_w, out_b, out, 1);
#undef RUN_GEMM
}
